// TorchRNNCell_58703613002087
// MI455X (gfx1250) — matrix operand order repaired and checked by reading, not compiled, not run on hardware
//
#include <hip/hip_runtime.h>
#include <hip/hip_bf16.h>
#include <math.h>

// Problem dims (fixed by the reference)
#define B_SZ  64
#define N_INP 512
#define N_HID 512
#define T_SZ  512

typedef __attribute__((ext_vector_type(16))) __bf16 v16bf;
typedef __attribute__((ext_vector_type(8)))  float  v8f;

// 32-byte WMMA 16-bit fragment viewed as two 16B global/LDS loads
union Frag16 {
    v16bf v;
    uint4 q[2];
};

// ---------------------------------------------------------------------------
// Prep: tiled transpose f32 -> bf16.
//   src: [rows, src_cols] f32 (row-major), reads coalesced along columns
//   dst[(c)*out_stride + r] = (bf16)src[r*src_cols + c]
// Used for X (per-batch [N_INP,T] slabs -> XT[t*B+b, i]) and for the weights
// (Wax/Waa -> WaxT/WaaT[h_out][k]) so WMMA fragment loads are contiguous b128.
// ---------------------------------------------------------------------------
__global__ __launch_bounds__(256) void transpose_to_bf16(
    const float* __restrict__ src, __bf16* __restrict__ dst,
    int src_cols, long out_stride, long src_batch_stride, long dst_batch_stride)
{
    __shared__ float tile[32][33];
    const int b  = blockIdx.z;
    src += (long)b * src_batch_stride;
    dst += (long)b * dst_batch_stride;
    const int r0 = blockIdx.y * 32, c0 = blockIdx.x * 32;
    const int tx = threadIdx.x, ty = threadIdx.y;
#pragma unroll
    for (int k = 0; k < 4; ++k) {
        const int r = ty + k * 8;
        tile[r][tx] = src[(long)(r0 + r) * src_cols + (c0 + tx)];
    }
    __syncthreads();
#pragma unroll
    for (int k = 0; k < 4; ++k) {
        const int c = ty + k * 8;
        dst[(long)(c0 + c) * out_stride + (r0 + tx)] = (__bf16)tile[tx][c];
    }
}

// ---------------------------------------------------------------------------
// Phase 1: U[t*B+b, h] = sum_i XT[t*B+b, i] * Wax[i, h] + bx[h] + ba[h]
// Written straight into d_out ([T,B,N_HID] f32). One 16x16 tile per wave,
// K = 512 -> 16x v_wmma_f32_16x16x32_bf16.
// ---------------------------------------------------------------------------
__global__ __launch_bounds__(256) void phase1_gemm(
    const __bf16* __restrict__ XT,    // [T*B, N_INP] bf16
    const __bf16* __restrict__ WaxT,  // [N_HID, N_INP] bf16 (transposed Wax)
    const float*  __restrict__ bx,
    const float*  __restrict__ ba,
    float* __restrict__ out)          // [T*B, N_HID] f32
{
    const int lane = threadIdx.x & 31;
    const int wave = threadIdx.x >> 5;
    const int rt   = blockIdx.x;               // 16-row tile of (t*B+b)
    const int ht   = blockIdx.y * 8 + wave;    // 16-col tile of h
    const int r0   = rt * 16, h0 = ht * 16;

    const int col = lane & 15;                 // A-row / B&C-col within tile
    const int hi  = lane >> 4;                 // lane half select

    // A (16x32 bf16): lanes 0-15 hold K {0-7,16-23}; lanes 16-31 hold {8-15,24-31}
    const __bf16* aptr = XT + (long)(r0 + col) * N_INP + (hi ? 8 : 0);
    // B (32x16 bf16): lanes 0-15 hold K {0-7,16-23}; lanes 16-31 hold {8-15,24-31} of column h0+col
    const __bf16* bptr = WaxT + (long)(h0 + col) * N_INP + hi * 8;

    v8f acc = {};
#pragma unroll
    for (int kc = 0; kc < 16; ++kc) {
        Frag16 a, bm;
        const __bf16* ap = aptr + kc * 32;
        a.q[0]  = *(const uint4*)(ap);
        a.q[1]  = *(const uint4*)(ap + 16);
        const __bf16* bp = bptr + kc * 32;
        bm.q[0] = *(const uint4*)(bp);
        bm.q[1] = *(const uint4*)(bp + 16);
        acc = __builtin_amdgcn_wmma_f32_16x16x32_bf16(
                  false, a.v, false, bm.v, (short)0, acc, false, false);
    }

    const float bias = bx[h0 + col] + ba[h0 + col];
#pragma unroll
    for (int m = 0; m < 8; ++m) {
        const int row = m + 8 * hi;            // C/D layout: lanes>=16 hold M=8..15
        out[(long)(r0 + row) * N_HID + (h0 + col)] = acc[m] + bias;
    }
}

// ---------------------------------------------------------------------------
// Phase 2: the sequential scan. 4 persistent workgroups (batch slices of 16),
// 32 waves each; wave j owns output columns [16j,16j+16). Waa B-fragments are
// preloaded into 128 VGPRs/lane; a_t lives in an LDS ping-pong buffer.
// Per step: 32 ds b128 loads (2 per fragment) + 16 WMMA + tanh + 1 barrier.
// ---------------------------------------------------------------------------
#define LDS_STRIDE 528   // bf16 elems per state row: 16B-aligned, bank-staggered

__global__ __launch_bounds__(1024) void phase2_rnn(
    const __bf16* __restrict__ WaaT,  // [N_HID, N_HID] bf16 (transposed Waa)
    float* __restrict__ out)          // [T, B, N_HID] f32, holds U, rewritten to A
{
    __shared__ __bf16 aS[2][16 * LDS_STRIDE];

    const int lane = threadIdx.x & 31;
    const int wave = threadIdx.x >> 5;     // 0..31 -> h tile
    const int b0   = blockIdx.x * 16;
    const int h0   = wave * 16;
    const int col  = lane & 15;
    const int hi   = lane >> 4;

    // Preload all Waa B-fragments for this wave's column tile (K = 0..511)
    Frag16 Bw[16];
    {
        const __bf16* bptr = WaaT + (long)(h0 + col) * N_HID + hi * 8;
#pragma unroll
        for (int kc = 0; kc < 16; ++kc) {
            const __bf16* bp = bptr + kc * 32;
            Bw[kc].q[0] = *(const uint4*)(bp);
            Bw[kc].q[1] = *(const uint4*)(bp + 16);
        }
    }

    // a_0 = 0
    for (int i = threadIdx.x; i < 16 * LDS_STRIDE; i += 1024)
        aS[0][i] = (__bf16)0.0f;
    __syncthreads();

#pragma unroll 1
    for (int t = 0; t < T_SZ; ++t) {
        const __bf16* abase = &aS[t & 1][col * LDS_STRIDE + (hi ? 8 : 0)];
        v8f acc = {};
#pragma unroll
        for (int kc = 0; kc < 16; ++kc) {
            Frag16 a;
            const __bf16* ap = abase + kc * 32;
            a.q[0] = *(const uint4*)(ap);
            a.q[1] = *(const uint4*)(ap + 16);
            acc = __builtin_amdgcn_wmma_f32_16x16x32_bf16(
                      false, a.v, false, Bw[kc].v, (short)0, acc, false, false);
        }

        // a_{t+1} = tanh(U_t + a_t @ Waa); write f32 to out, bf16 to LDS ping
        const long tbase = (long)t * (B_SZ * N_HID) + (long)b0 * N_HID;
#pragma unroll
        for (int m = 0; m < 8; ++m) {
            const int  row  = m + 8 * hi;
            const long gidx = tbase + (long)row * N_HID + (h0 + col);
            const float v   = tanhf(acc[m] + out[gidx]);
            out[gidx] = v;
            aS[(t + 1) & 1][row * LDS_STRIDE + (h0 + col)] = (__bf16)v;
        }
        __syncthreads();
    }
}

// ---------------------------------------------------------------------------
extern "C" void kernel_launch(void* const* d_in, const int* in_sizes, int n_in,
                              void* d_out, int out_size, void* d_ws, size_t ws_size,
                              hipStream_t stream)
{
    const float* X   = (const float*)d_in[0];  // [B, N_INP, T]
    const float* Wax = (const float*)d_in[1];  // [N_INP, N_HID]
    const float* Waa = (const float*)d_in[2];  // [N_HID, N_HID]
    const float* bx  = (const float*)d_in[3];  // [N_HID]
    const float* ba  = (const float*)d_in[4];  // [N_HID]
    float* out = (float*)d_out;                // [T, B, N_HID]

    // Workspace layout (bf16): XT [T*B, N_INP], WaxT [N_HID, N_INP], WaaT [N_HID, N_HID]
    __bf16* XT   = (__bf16*)d_ws;
    __bf16* WaxT = XT   + (size_t)T_SZ * B_SZ * N_INP;
    __bf16* WaaT = WaxT + (size_t)N_HID * N_INP;

    const dim3 blkT(32, 8);
    // X: per-b slab [N_INP, T] -> XT[(t*B+b), i]
    transpose_to_bf16<<<dim3(T_SZ / 32, N_INP / 32, B_SZ), blkT, 0, stream>>>(
        X, XT, T_SZ, (long)B_SZ * N_INP, (long)N_INP * T_SZ, (long)N_INP);
    // Wax [N_INP, N_HID] -> WaxT[h][i]
    transpose_to_bf16<<<dim3(N_HID / 32, N_INP / 32, 1), blkT, 0, stream>>>(
        Wax, WaxT, N_HID, (long)N_INP, 0L, 0L);
    // Waa [N_HID, N_HID] -> WaaT[h_out][h_in]
    transpose_to_bf16<<<dim3(N_HID / 32, N_HID / 32, 1), blkT, 0, stream>>>(
        Waa, WaaT, N_HID, (long)N_HID, 0L, 0L);

    // Phase 1: U into d_out. 2048 row-tiles x 32 col-tiles, 8 waves/block.
    phase1_gemm<<<dim3((T_SZ * B_SZ) / 16, 4), dim3(256), 0, stream>>>(
        XT, WaxT, bx, ba, out);

    // Phase 2: 4 persistent workgroups scan T=512 steps.
    phase2_rnn<<<dim3(B_SZ / 16), dim3(1024), 0, stream>>>(WaaT, out);
}
